// LocalAttention_9191230013574
// MI455X (gfx1250) — matrix operand order repaired and checked by reading, not compiled, not run on hardware
//
#include <hip/hip_runtime.h>

typedef __attribute__((ext_vector_type(16))) _Float16 v16h;
typedef __attribute__((ext_vector_type(4)))  _Float16 v4h;
typedef __attribute__((ext_vector_type(8)))  float    v8f;

union F16Frag { v16h h; uint4 u[2]; };

#define SEQ 8192
#define DIM 64
#define WIN 128
#define JW  256   // keys per window (look_back 1 + current)

// LDS layout (bytes). RoPE table is dead after staging, so P overlays it.
#define OFF_TAB 0                            // float cos/sin [256][32][2] = 64 KB
#define OFF_P   0                            // f16 P [128][256]          = 64 KB
#define OFF_Q   65536                        // f16 Q [128][64]           = 16 KB
#define OFF_K   (65536 + 16384)              // f16 K [256][64]           = 32 KB
#define OFF_VT  (65536 + 16384 + 32768)      // f16 V^T [64][256]         = 32 KB
#define SMEM_BYTES (65536 + 16384 + 32768 + 32768)   // 144 KB

static __device__ __forceinline__ v8f wmma_f16(v16h a, v16h b, v8f c) {
  // D = A(16x32 f16) * B(32x16 f16) + C(16x16 f32)
  return __builtin_amdgcn_wmma_f32_16x16x32_f16(false, a, false, b, (short)0, c,
                                                false, false);
}

__global__ __launch_bounds__(256, 1)
void local_attn_rope_kernel(const float* __restrict__ q,
                            const float* __restrict__ k,
                            const float* __restrict__ v,
                            float* __restrict__ out)
{
  extern __shared__ char smem[];
  float*    tab = (float*)(smem + OFF_TAB);
  _Float16* Ph  = (_Float16*)(smem + OFF_P);
  _Float16* Qh  = (_Float16*)(smem + OFF_Q);
  _Float16* Kh  = (_Float16*)(smem + OFF_K);
  _Float16* Vt  = (_Float16*)(smem + OFF_VT);

  const int tid = threadIdx.x;
  const int bb  = blockIdx.x >> 6;   // batch
  const int ww  = blockIdx.x & 63;   // window

  // ---------- Phase 0: RoPE table tab[(p*32+f)*2] = {cos,sin}(p * 10000^(-f/32))
  {
    const float f    = (float)(tid & 31);                       // freq idx fixed per thread
    const float invf = __expf(f * (-9.210340371976184f / 32.0f)); // 10000^(-f/32)
    #pragma unroll
    for (int t = 0; t < 32; ++t) {
      const int   idx = tid + (t << 8);       // idx = p*32 + f
      const float ang = (float)(idx >> 5) * invf;
      float sv, cv;
      __sincosf(ang, &sv, &cv);
      tab[idx * 2 + 0] = cv;
      tab[idx * 2 + 1] = sv;
    }
  }
  __syncthreads();

  // ---------- Phase 1: stage Q (scale+RoPE), K (RoPE), V (transposed) as f16
  const float  scale = 0.125f;  // 64^-0.5
  const float* qg = q + ((size_t)bb * SEQ + (size_t)ww * WIN) * DIM;
  const float* kg = k + (size_t)bb * SEQ * DIM + (long)(ww - 1) * WIN * DIM;
  const float* vg = v + (size_t)bb * SEQ * DIM + (long)(ww - 1) * WIN * DIM;

  #pragma unroll
  for (int t = 0; t < 8; ++t) {               // Q: 128 rows x 16 float4 segs
    const int idx = tid + (t << 8);
    const int row = idx >> 4;
    const int d0  = (idx & 15) << 2;
    const float4 x = *(const float4*)(qg + (size_t)row * DIM + d0);
    const float4 y = *(const float4*)(qg + (size_t)row * DIM + (d0 ^ 32));
    const float  sgn = (d0 < 32) ? -1.0f : 1.0f;
    const float* cs = tab + ((((128 + row) << 5) + (d0 & 31)) << 1);
    v4h hv;
    hv.x = (_Float16)(scale * (x.x * cs[0] + sgn * y.x * cs[1]));
    hv.y = (_Float16)(scale * (x.y * cs[2] + sgn * y.y * cs[3]));
    hv.z = (_Float16)(scale * (x.z * cs[4] + sgn * y.z * cs[5]));
    hv.w = (_Float16)(scale * (x.w * cs[6] + sgn * y.w * cs[7]));
    *(v4h*)(Qh + row * DIM + d0) = hv;
  }

  #pragma unroll
  for (int t = 0; t < 16; ++t) {              // K: 256 rows x 16 float4 segs
    const int idx = tid + (t << 8);
    const int c   = idx >> 4;
    const int d0  = (idx & 15) << 2;
    float4 x = make_float4(0.f, 0.f, 0.f, 0.f);
    float4 y = make_float4(0.f, 0.f, 0.f, 0.f);
    if (ww != 0 || c >= 128) {                // w==0 look-back is pad (fully masked)
      x = *(const float4*)(kg + (size_t)c * DIM + d0);
      y = *(const float4*)(kg + (size_t)c * DIM + (d0 ^ 32));
    }
    const float  sgn = (d0 < 32) ? -1.0f : 1.0f;
    const float* cs = tab + (((c << 5) + (d0 & 31)) << 1);
    v4h hv;
    hv.x = (_Float16)(x.x * cs[0] + sgn * y.x * cs[1]);
    hv.y = (_Float16)(x.y * cs[2] + sgn * y.y * cs[3]);
    hv.z = (_Float16)(x.z * cs[4] + sgn * y.z * cs[5]);
    hv.w = (_Float16)(x.w * cs[6] + sgn * y.w * cs[7]);
    *(v4h*)(Kh + c * DIM + d0) = hv;
  }

  #pragma unroll
  for (int t = 0; t < 16; ++t) {              // V: transpose into Vt[d][c]
    const int idx = tid + (t << 8);
    const int c   = idx >> 4;
    const int d0  = (idx & 15) << 2;
    float4 x = make_float4(0.f, 0.f, 0.f, 0.f);
    if (ww != 0 || c >= 128) x = *(const float4*)(vg + (size_t)c * DIM + d0);
    Vt[(d0 + 0) * JW + c] = (_Float16)x.x;
    Vt[(d0 + 1) * JW + c] = (_Float16)x.y;
    Vt[(d0 + 2) * JW + c] = (_Float16)x.z;
    Vt[(d0 + 3) * JW + c] = (_Float16)x.w;
  }
  __syncthreads();

  // ---------- Phase 2: per-wave 16-query strip
  const int wave  = tid >> 5;
  const int lane  = tid & 31;
  const int g     = lane >> 4;     // half-wave group
  const int n     = lane & 15;     // N index (cols of tile)
  const int qbase = wave << 4;
  const int arow  = qbase + n;     // A-fragment row (M = lane%16)

  // Q A-fragments per ISA 16-bit A layout: K = (t<8 ? t : t+8) + g*8 (+d0)
  F16Frag a0, a1;
  a0.u[0] = *(const uint4*)(Qh + arow * DIM + 0  + g * 8);
  a0.u[1] = *(const uint4*)(Qh + arow * DIM + 16 + g * 8);
  a1.u[0] = *(const uint4*)(Qh + arow * DIM + 32 + g * 8);
  a1.u[1] = *(const uint4*)(Qh + arow * DIM + 48 + g * 8);

  const int ct_lo = (ww == 0) ? 8 : 0;   // pad-masked tiles skipped
  const int ct_hi = wave + 8;            // causal: tiles beyond diagonal skipped

  v8f   S[16] = {};
  float m[8], sum[8];
  #pragma unroll
  for (int r = 0; r < 8; ++r) { m[r] = -3.0e38f; sum[r] = 0.0f; }

  #pragma unroll
  for (int ct = 0; ct < 16; ++ct) {
    if (ct >= ct_lo && ct <= ct_hi) {
      const int c = (ct << 4) + n;       // key column for this lane
      // B in same K order as A: K = (t<8 ? t : t+8) + g*8 (+d0)
      F16Frag b0, b1;
      b0.u[0] = *(const uint4*)(Kh + c * DIM + 0  + g * 8);
      b0.u[1] = *(const uint4*)(Kh + c * DIM + 16 + g * 8);
      b1.u[0] = *(const uint4*)(Kh + c * DIM + 32 + g * 8);
      b1.u[1] = *(const uint4*)(Kh + c * DIM + 48 + g * 8);
      v8f acc = {};
      acc = wmma_f16(a0.h, b0.h, acc);
      acc = wmma_f16(a1.h, b1.h, acc);
      if (ct == ct_hi) {                 // partial causal mask on diagonal tile
        #pragma unroll
        for (int r = 0; r < 8; ++r) {
          const int i = qbase + r + (g << 3);
          if (c > i + 128) acc[r] = -3.0e38f;
        }
      }
      #pragma unroll
      for (int r = 0; r < 8; ++r) m[r] = fmaxf(m[r], acc[r]);
      S[ct] = acc;
    }
  }

  // row max across the 16 lanes holding one row (xor<16 never crosses half-wave)
  #pragma unroll
  for (int r = 0; r < 8; ++r) {
    float mm = m[r];
    mm = fmaxf(mm, __shfl_xor(mm, 8, 32));
    mm = fmaxf(mm, __shfl_xor(mm, 4, 32));
    mm = fmaxf(mm, __shfl_xor(mm, 2, 32));
    mm = fmaxf(mm, __shfl_xor(mm, 1, 32));
    m[r] = mm;
  }

  #pragma unroll
  for (int ct = 0; ct < 16; ++ct) {
    if (ct >= ct_lo && ct <= ct_hi) {
      #pragma unroll
      for (int r = 0; r < 8; ++r) {
        const float p = __expf(S[ct][r] - m[r]);   // masked -> exp(-huge) = 0
        S[ct][r] = p;
        sum[r] += p;
      }
    }
  }
  #pragma unroll
  for (int r = 0; r < 8; ++r) {
    float ss = sum[r];
    ss += __shfl_xor(ss, 8, 32);
    ss += __shfl_xor(ss, 4, 32);
    ss += __shfl_xor(ss, 2, 32);
    ss += __shfl_xor(ss, 1, 32);
    sum[r] = 1.0f / ss;
  }

  // store normalized P as f16 (zeros for skipped tiles) — C-layout -> LDS row-major
  #pragma unroll
  for (int ct = 0; ct < 16; ++ct) {
    const int  c    = (ct << 4) + n;
    const bool live = (ct >= ct_lo && ct <= ct_hi);
    #pragma unroll
    for (int r = 0; r < 8; ++r) {
      const int i = qbase + r + (g << 3);
      const float val = live ? S[ct][r] * sum[r] : 0.0f;
      Ph[i * JW + c] = (_Float16)val;
    }
  }

  // ---------- Phase 3: O = P(16x256) * V(256x64), K-chunks of 32
  const int kk_lo = (ww == 0) ? 4 : 0;
  const int kk_hi = (qbase + 143) >> 5;   // last chunk with unmasked keys (<=7)
  float* outg = out + ((size_t)bb * SEQ + (size_t)ww * WIN) * DIM;

  #pragma unroll
  for (int oc = 0; oc < 4; ++oc) {
    v8f o = {};
    for (int kk = kk_lo; kk <= kk_hi; ++kk) {
      const int kb = kk << 5;
      F16Frag pa, vb;
      pa.u[0] = *(const uint4*)(Ph + arow * JW + kb + g * 8);
      pa.u[1] = *(const uint4*)(Ph + arow * JW + kb + 16 + g * 8);
      const int vr = (oc << 4) + n;
      vb.u[0] = *(const uint4*)(Vt + vr * JW + kb + g * 8);
      vb.u[1] = *(const uint4*)(Vt + vr * JW + kb + 16 + g * 8);
      o = wmma_f16(pa.h, vb.h, o);
    }
    #pragma unroll
    for (int r = 0; r < 8; ++r) {
      const int i = qbase + r + (g << 3);
      outg[(size_t)i * DIM + (oc << 4) + n] = o[r];
    }
  }
}

extern "C" void kernel_launch(void* const* d_in, const int* in_sizes, int n_in,
                              void* d_out, int out_size, void* d_ws, size_t ws_size,
                              hipStream_t stream) {
  (void)in_sizes; (void)n_in; (void)out_size; (void)d_ws; (void)ws_size;
  const float* q = (const float*)d_in[0];
  const float* k = (const float*)d_in[1];
  const float* v = (const float*)d_in[2];
  float* out = (float*)d_out;

  (void)hipFuncSetAttribute((const void*)local_attn_rope_kernel,
                            hipFuncAttributeMaxDynamicSharedMemorySize,
                            SMEM_BYTES);
  dim3 grid(16 * 64);   // one block per (batch, window)
  dim3 block(256);      // 8 waves: one 16-query strip each
  local_attn_rope_kernel<<<grid, block, SMEM_BYTES, stream>>>(q, k, v, out);
}
